// GFNet_12283606466607
// MI455X (gfx1250) — matrix operand order repaired and checked by reading, not compiled, not run on hardware
//
#include <hip/hip_runtime.h>
#include <hip/hip_bf16.h>

#define B_   4
#define C_   128
#define HW_  4096
#define EPS_ 1e-5f
#define OUT_SCALE 4096.0f

typedef __attribute__((ext_vector_type(16))) _Float16 v16h;
typedef __attribute__((ext_vector_type(8)))  _Float16 v8h;
typedef __attribute__((ext_vector_type(8)))  float    v8f;

__device__ __forceinline__ int lane_id() { return threadIdx.x & 31; }

// A fragment 16x32 f16: element (m,k) = base[m*stride + k]
// ISA layout: lane l holds M=l&15; k-offsets (l>>4)*8 .. +7 and +16 .. +23 (two 16B loads)
__device__ __forceinline__ v16h load_a_frag(const _Float16* base, int stride) {
  const int l = lane_id();
  const int m = l & 15;
  const int kb = (l >> 4) * 8;
  const _Float16* p = base + (size_t)m * stride + kb;
  v8h lo = *(const v8h*)(p);
  v8h hi = *(const v8h*)(p + 16);
  v16h r;
#pragma unroll
  for (int i = 0; i < 8; ++i) { r[i] = lo[i]; r[i + 8] = hi[i]; }
  return r;
}

// B fragment 32x16 f16: element (k,n) = base[n*stride + k]
// ISA layout: lane l holds N=l&15; k-offsets (l>>4)*8 .. +7 and +16 .. +23 (two 16B loads)
__device__ __forceinline__ v16h load_b_frag(const _Float16* base, int stride) {
  const int l = lane_id();
  const int n = l & 15;
  const int kb = (l >> 4) * 8;
  const _Float16* p = base + (size_t)n * stride + kb;
  v8h lo = *(const v8h*)(p);
  v8h hi = *(const v8h*)(p + 16);
  v16h r;
#pragma unroll
  for (int i = 0; i < 8; ++i) { r[i] = lo[i]; r[i + 8] = hi[i]; }
  return r;
}

// ---------------- per-(b,c) mean / rstd over spatial dims (ddof=1) ----------------
__global__ void mvn_stats_kernel(const float* __restrict__ content,
                                 const float* __restrict__ guidance,
                                 float* __restrict__ meanC, float* __restrict__ rstdC,
                                 float* __restrict__ meanG, float* __restrict__ rstdG) {
  const int row   = blockIdx.x & 511;   // b*C + c
  const int which = blockIdx.x >> 9;    // 0=content 1=guidance
  const float* src = (which ? guidance : content) + (size_t)row * HW_;
  float s = 0.f, sq = 0.f;
  for (int i = threadIdx.x; i < HW_; i += 256) { float v = src[i]; s += v; sq += v * v; }
  __shared__ float shs[256], shq[256];
  shs[threadIdx.x] = s; shq[threadIdx.x] = sq;
  __syncthreads();
  for (int off = 128; off > 0; off >>= 1) {
    if (threadIdx.x < off) { shs[threadIdx.x] += shs[threadIdx.x + off]; shq[threadIdx.x] += shq[threadIdx.x + off]; }
    __syncthreads();
  }
  if (threadIdx.x == 0) {
    float mean = shs[0] * (1.0f / HW_);
    float var  = (shq[0] - shs[0] * mean) * (1.0f / (HW_ - 1)) + EPS_;
    float rs   = rsqrtf(var);
    if (which) { meanG[row] = mean; rstdG[row] = rs; }
    else       { meanC[row] = mean; rstdC[row] = rs; }
  }
}

// ---------- pack normalized/raw activations to f16 in [b][hw][c] layout ----------
__global__ void pack_kernel(const float* __restrict__ content, const float* __restrict__ guidance,
                            const float* __restrict__ meanC, const float* __restrict__ rstdC,
                            const float* __restrict__ meanG, const float* __restrict__ rstdG,
                            _Float16* __restrict__ Cn, _Float16* __restrict__ Gn, _Float16* __restrict__ Gr) {
  const int gid = blockIdx.x * blockDim.x + threadIdx.x;  // b*HW + p
  if (gid >= B_ * HW_) return;
  const int b = gid >> 12;
  const int p = gid & (HW_ - 1);
#pragma unroll 4
  for (int c = 0; c < C_; ++c) {
    const size_t src = ((size_t)(b * C_ + c)) * HW_ + p;   // coalesced over p
    const float cv = content[src], gv = guidance[src];
    Cn[(size_t)gid * C_ + c] = (_Float16)((cv - meanC[b * C_ + c]) * rstdC[b * C_ + c]);
    Gn[(size_t)gid * C_ + c] = (_Float16)((gv - meanG[b * C_ + c]) * rstdG[b * C_ + c]);
    Gr[(size_t)gid * C_ + c] = (_Float16)gv;
  }
}

// ---------------- convert the four [C,C] weight matrices to f16 ----------------
__global__ void wcvt_kernel(const float* __restrict__ w0, const float* __restrict__ w1,
                            const float* __restrict__ w2, const float* __restrict__ w3,
                            _Float16* __restrict__ out) {
  const int idx = blockIdx.x * blockDim.x + threadIdx.x;  // < 4*16384
  const float* src = (idx < 16384) ? w0 : (idx < 32768) ? w1 : (idx < 49152) ? w2 : w3;
  out[idx] = (_Float16)src[idx & 16383];
}

// ---------------- 1x1 conv as WMMA GEMM: Y[o,p] = alpha*sum_c W[o,c]X[c,p] + bias[o] ----
// X in [b][hw][c] f16.  Optional outputs: f32 [b][c][hw], f16 [b][hw][c], f16 [b][c][hw].
__global__ void __launch_bounds__(128) conv_gemm_kernel(
    const _Float16* __restrict__ W, const _Float16* __restrict__ Xpc,
    const float* __restrict__ bias, float alpha,
    float* __restrict__ Ycp, _Float16* __restrict__ Ypc, _Float16* __restrict__ Ycp16) {
  const int b  = blockIdx.y;
  const int p0 = blockIdx.x * 64;
  const int wv = threadIdx.x >> 5;
  const int l  = threadIdx.x & 31;
  const int hl = l >> 4, ln = l & 15;
  const int orow = wv * 32;
  const _Float16* Xb = Xpc + (size_t)b * HW_ * C_;

  const v8f vzero = {};
  v8f acc[2][4];
#pragma unroll
  for (int i = 0; i < 2; ++i)
#pragma unroll
    for (int j = 0; j < 4; ++j) acc[i][j] = vzero;

#pragma unroll
  for (int kk = 0; kk < 4; ++kk) {
    v16h a0 = load_a_frag(W + (size_t)orow * C_ + kk * 32, C_);
    v16h a1 = load_a_frag(W + (size_t)(orow + 16) * C_ + kk * 32, C_);
#pragma unroll
    for (int j = 0; j < 4; ++j) {
      v16h bf = load_b_frag(Xb + (size_t)(p0 + j * 16) * C_ + kk * 32, C_);
      acc[0][j] = __builtin_amdgcn_wmma_f32_16x16x32_f16(false, a0, false, bf, (short)0, acc[0][j], false, false);
      acc[1][j] = __builtin_amdgcn_wmma_f32_16x16x32_f16(false, a1, false, bf, (short)0, acc[1][j], false, false);
    }
  }

#pragma unroll
  for (int i = 0; i < 2; ++i) {
    const int ob = orow + i * 16 + hl * 8;      // D layout: row = r + 8*(lane>>4)
    float bv[8];
#pragma unroll
    for (int r = 0; r < 8; ++r) bv[r] = bias[ob + r];
#pragma unroll
    for (int j = 0; j < 4; ++j) {
      const int p = p0 + j * 16 + ln;
      float vals[8];
#pragma unroll
      for (int r = 0; r < 8; ++r) vals[r] = acc[i][j][r] * alpha + bv[r];
      if (Ycp) {
        float* dst = Ycp + ((size_t)b * C_ + ob) * HW_ + p;
#pragma unroll
        for (int r = 0; r < 8; ++r) dst[(size_t)r * HW_] = vals[r];
      }
      if (Ycp16) {
        _Float16* dst = Ycp16 + ((size_t)b * C_ + ob) * HW_ + p;
#pragma unroll
        for (int r = 0; r < 8; ++r) dst[(size_t)r * HW_] = (_Float16)vals[r];
      }
      if (Ypc) {
        v8h hv;
#pragma unroll
        for (int r = 0; r < 8; ++r) hv[r] = (_Float16)vals[r];
        *(v8h*)(Ypc + ((size_t)b * HW_ + p) * C_ + ob) = hv;
      }
    }
  }
}

// -------- fused flash attention: G2pre[b][c][n] = sum_m softmax_m(Q[n]·K[m]) V[c,m] ----
__global__ void __launch_bounds__(128) attention_kernel(
    const _Float16* __restrict__ Qpc,   // [b][hw][c]  (F)
    const _Float16* __restrict__ Kpc,   // [b][hw][c]  (G1)
    const _Float16* __restrict__ Vcp,   // [b][c][hw]  (H)
    float* __restrict__ G2pre) {        // [b][c][hw]
  const int b  = blockIdx.y;
  const int wv = threadIdx.x >> 5;
  const int l  = threadIdx.x & 31;
  const int hl = l >> 4, ln = l & 15;
  const int q0 = blockIdx.x * 64 + wv * 16;

  const _Float16* Qb = Qpc + (size_t)b * HW_ * C_;
  const _Float16* Kb = Kpc + (size_t)b * HW_ * C_;
  const _Float16* Vb = Vcp + (size_t)b * C_ * HW_;

  __shared__ _Float16 ldsP[4][16 * 64];   // per-wave 16x64 P staging (C->A relayout)
  _Float16* myP = &ldsP[wv][0];

  v16h qa[4];
#pragma unroll
  for (int kk = 0; kk < 4; ++kk) qa[kk] = load_a_frag(Qb + (size_t)q0 * C_ + kk * 32, C_);

  const v8f vzero = {};
  v8f oacc[8];
#pragma unroll
  for (int i = 0; i < 8; ++i) oacc[i] = vzero;
  float mrow[8], lrow[8];
#pragma unroll
  for (int r = 0; r < 8; ++r) { mrow[r] = -3.0e38f; lrow[r] = 0.f; }

  for (int kt = 0; kt < HW_ / 64; ++kt) {
    const int kbase = kt * 64;
    v8f s[4];
#pragma unroll
    for (int j = 0; j < 4; ++j) s[j] = vzero;
#pragma unroll
    for (int kk = 0; kk < 4; ++kk) {
#pragma unroll
      for (int j = 0; j < 4; ++j) {
        v16h bf = load_b_frag(Kb + (size_t)(kbase + j * 16) * C_ + kk * 32, C_);
        s[j] = __builtin_amdgcn_wmma_f32_16x16x32_f16(false, qa[kk], false, bf, (short)0, s[j], false, false);
      }
    }
    // row max over 64 keys (reduce across the 16-lane half-groups)
    float tmax[8], scale[8], tsum[8];
#pragma unroll
    for (int r = 0; r < 8; ++r) {
      float v = fmaxf(fmaxf(s[0][r], s[1][r]), fmaxf(s[2][r], s[3][r]));
      v = fmaxf(v, __shfl_xor(v, 1, 32));
      v = fmaxf(v, __shfl_xor(v, 2, 32));
      v = fmaxf(v, __shfl_xor(v, 4, 32));
      v = fmaxf(v, __shfl_xor(v, 8, 32));
      tmax[r] = v;
    }
#pragma unroll
    for (int r = 0; r < 8; ++r) {
      const float nm = fmaxf(mrow[r], tmax[r]);
      scale[r] = __expf(mrow[r] - nm);
      mrow[r]  = nm;
      tsum[r]  = 0.f;
    }
    // P = exp(S - m): write to LDS in A-operand row-major [m][64]
#pragma unroll
    for (int j = 0; j < 4; ++j) {
#pragma unroll
      for (int r = 0; r < 8; ++r) {
        const float pe = __expf(s[j][r] - mrow[r]);
        tsum[r] += pe;
        myP[(r + hl * 8) * 64 + j * 16 + ln] = (_Float16)pe;
      }
    }
#pragma unroll
    for (int r = 0; r < 8; ++r) {
      float v = tsum[r];
      v += __shfl_xor(v, 1, 32);
      v += __shfl_xor(v, 2, 32);
      v += __shfl_xor(v, 4, 32);
      v += __shfl_xor(v, 8, 32);
      lrow[r] = lrow[r] * scale[r] + v;
    }
#pragma unroll
    for (int i = 0; i < 8; ++i)
#pragma unroll
      for (int r = 0; r < 8; ++r) oacc[i][r] *= scale[r];

    __syncthreads();
    v16h pa0 = load_a_frag(myP, 64);
    v16h pa1 = load_a_frag(myP + 32, 64);
#pragma unroll
    for (int ct = 0; ct < 8; ++ct) {
      v16h b0 = load_b_frag(Vb + (size_t)(ct * 16) * HW_ + kbase, HW_);
      v16h b1 = load_b_frag(Vb + (size_t)(ct * 16) * HW_ + kbase + 32, HW_);
      oacc[ct] = __builtin_amdgcn_wmma_f32_16x16x32_f16(false, pa0, false, b0, (short)0, oacc[ct], false, false);
      oacc[ct] = __builtin_amdgcn_wmma_f32_16x16x32_f16(false, pa1, false, b1, (short)0, oacc[ct], false, false);
    }
    __syncthreads();
  }

  float inv[8];
#pragma unroll
  for (int r = 0; r < 8; ++r) inv[r] = 1.0f / lrow[r];
#pragma unroll
  for (int ct = 0; ct < 8; ++ct) {
    const int c = ct * 16 + ln;
    float* dst = G2pre + ((size_t)b * C_ + c) * HW_ + q0 + hl * 8;
    float4 v0 = make_float4(oacc[ct][0] * inv[0], oacc[ct][1] * inv[1], oacc[ct][2] * inv[2], oacc[ct][3] * inv[3]);
    float4 v1 = make_float4(oacc[ct][4] * inv[4], oacc[ct][5] * inv[5], oacc[ct][6] * inv[6], oacc[ct][7] * inv[7]);
    *(float4*)dst       = v0;
    *(float4*)(dst + 4) = v1;
  }
}

// ---------------- second softmax: over hw axis, per (b,c), in place ----------------
__global__ void softmax_hw_kernel(float* __restrict__ G2) {
  float* x = G2 + (size_t)blockIdx.x * HW_;
  __shared__ float sh[256];
  float m = -3.0e38f;
  for (int i = threadIdx.x; i < HW_; i += 256) m = fmaxf(m, x[i]);
  sh[threadIdx.x] = m; __syncthreads();
  for (int off = 128; off > 0; off >>= 1) { if (threadIdx.x < off) sh[threadIdx.x] = fmaxf(sh[threadIdx.x], sh[threadIdx.x + off]); __syncthreads(); }
  m = sh[0]; __syncthreads();
  float s = 0.f;
  for (int i = threadIdx.x; i < HW_; i += 256) s += __expf(x[i] - m);
  sh[threadIdx.x] = s; __syncthreads();
  for (int off = 128; off > 0; off >>= 1) { if (threadIdx.x < off) sh[threadIdx.x] += sh[threadIdx.x + off]; __syncthreads(); }
  const float inv = 1.0f / sh[0];
  for (int i = threadIdx.x; i < HW_; i += 256) x[i] = __expf(x[i] - m) * inv;
}

// ---------------- gamma/beta: channel-weighted sums of G2 ----------------
__global__ void gamma_beta_kernel(const float* __restrict__ G2,
                                  const float* __restrict__ gw, const float* __restrict__ bw,
                                  float* __restrict__ gamma, float* __restrict__ beta) {
  const int idx = blockIdx.x * blockDim.x + threadIdx.x;  // b*HW + q
  if (idx >= B_ * HW_) return;
  const int b = idx >> 12, q = idx & (HW_ - 1);
  float ag = 0.f, ab = 0.f;
  for (int c = 0; c < C_; ++c) {
    const float v = G2[((size_t)b * C_ + c) * HW_ + q];
    ag += v * gw[c]; ab += v * bw[c];
  }
  gamma[idx] = ag; beta[idx] = ab;
}

// ------ Out[b,c] = G1[b,c](64x64) @ gammaM[b](64x64) + betaM[b]; scaled -> f16 [hw][c]
__global__ void out_matmul_kernel(const float* __restrict__ G1cp,
                                  const float* __restrict__ gamma,
                                  const float* __restrict__ beta,
                                  _Float16* __restrict__ Out16pc) {
  const int bc = blockIdx.x;                 // b*C + c
  const int b = bc >> 7, c = bc & 127;
  __shared__ float sG[64 * 64];
  for (int i = threadIdx.x; i < 4096; i += 256) sG[i] = gamma[(size_t)b * HW_ + i];
  __syncthreads();
  const float* g1 = G1cp + (size_t)bc * HW_;
  for (int t = threadIdx.x; t < 4096; t += 256) {
    const int i = t >> 6, j = t & 63;
    float acc = 0.f;
#pragma unroll 8
    for (int k = 0; k < 64; ++k) acc += g1[i * 64 + k] * sG[k * 64 + j];
    const float out = (acc + beta[(size_t)b * HW_ + t]) * OUT_SCALE;
    Out16pc[((size_t)b * HW_ + t) * C_ + c] = (_Float16)out;
  }
}

extern "C" void kernel_launch(void* const* d_in, const int* in_sizes, int n_in,
                              void* d_out, int out_size, void* d_ws, size_t ws_size,
                              hipStream_t stream) {
  (void)in_sizes; (void)n_in; (void)out_size; (void)ws_size;
  const float* content  = (const float*)d_in[0];
  const float* guidance = (const float*)d_in[1];
  const float* g1_w = (const float*)d_in[2];
  const float* g1_b = (const float*)d_in[3];
  const float* f_w  = (const float*)d_in[4];
  const float* f_b  = (const float*)d_in[5];
  const float* h_w  = (const float*)d_in[6];
  const float* h_b  = (const float*)d_in[7];
  const float* gamma_w = (const float*)d_in[8];
  const float* beta_w  = (const float*)d_in[9];
  const float* out_w   = (const float*)d_in[10];
  const float* out_b   = (const float*)d_in[11];

  char* ws = (char*)d_ws;
  size_t off = 0;
  auto alloc = [&](size_t bytes) -> void* {
    void* p = ws + off;
    off = (off + bytes + 255) & ~(size_t)255;
    return p;
  };

  const size_t ACT16 = (size_t)B_ * HW_ * C_ * sizeof(_Float16);  // 4 MB
  const size_t ACT32 = (size_t)B_ * HW_ * C_ * sizeof(float);     // 8 MB

  float*    meanC = (float*)alloc(512 * 4);
  float*    rstdC = (float*)alloc(512 * 4);
  float*    meanG = (float*)alloc(512 * 4);
  float*    rstdG = (float*)alloc(512 * 4);
  _Float16* Cn    = (_Float16*)alloc(ACT16);
  _Float16* Gn    = (_Float16*)alloc(ACT16);
  _Float16* Gr    = (_Float16*)alloc(ACT16);
  _Float16* W16   = (_Float16*)alloc(4 * (size_t)C_ * C_ * sizeof(_Float16));
  float*    G1cp  = (float*)alloc(ACT32);
  _Float16* Kpc   = (_Float16*)alloc(ACT16);
  _Float16* Qpc   = (_Float16*)alloc(ACT16);
  _Float16* Vcp   = (_Float16*)alloc(ACT16);
  float*    G2    = (float*)alloc(ACT32);
  float*    gammaA = (float*)alloc((size_t)B_ * HW_ * 4);
  float*    betaA  = (float*)alloc((size_t)B_ * HW_ * 4);
  _Float16* Out16  = (_Float16*)alloc(ACT16);

  mvn_stats_kernel<<<1024, 256, 0, stream>>>(content, guidance, meanC, rstdC, meanG, rstdG);
  pack_kernel<<<64, 256, 0, stream>>>(content, guidance, meanC, rstdC, meanG, rstdG, Cn, Gn, Gr);
  wcvt_kernel<<<256, 256, 0, stream>>>(g1_w, f_w, h_w, out_w, W16);

  dim3 cg(HW_ / 64, B_);
  // G1 = g1_w @ mvn(content): f32 [c][hw] (for Out matmul) + f16 [hw][c] (K)
  conv_gemm_kernel<<<cg, 128, 0, stream>>>(W16 + 0 * 16384, Cn, g1_b, 1.0f, G1cp, Kpc, nullptr);
  // F = f_w @ mvn(guidance): f16 [hw][c] (Q)
  conv_gemm_kernel<<<cg, 128, 0, stream>>>(W16 + 1 * 16384, Gn, f_b, 1.0f, nullptr, Qpc, nullptr);
  // H = h_w @ guidance: f16 [c][hw] (V)
  conv_gemm_kernel<<<cg, 128, 0, stream>>>(W16 + 2 * 16384, Gr, h_b, 1.0f, nullptr, nullptr, Vcp);

  attention_kernel<<<cg, 128, 0, stream>>>(Qpc, Kpc, Vcp, G2);
  softmax_hw_kernel<<<512, 256, 0, stream>>>(G2);
  gamma_beta_kernel<<<64, 256, 0, stream>>>(G2, gamma_w, beta_w, gammaA, betaA);
  out_matmul_kernel<<<512, 256, 0, stream>>>(G1cp, gammaA, betaA, Out16);

  // final = out_w @ Out + out_b; undo the 4096x scaling via alpha
  conv_gemm_kernel<<<cg, 128, 0, stream>>>(W16 + 3 * 16384, Out16, out_b, 1.0f / OUT_SCALE,
                                           (float*)d_out, nullptr, nullptr);
}
